// LSTM_GCC_Decoder_80178449482106
// MI455X (gfx1250) — matrix operand order repaired and checked by reading, not compiled, not run on hardware
//
#include <hip/hip_runtime.h>
#include <hip/hip_bf16.h>
#include <stdint.h>
#include <stddef.h>

// Problem dimensions (fixed by the reference)
#define H_    512
#define MID_  256
#define M_    2
#define B_    128
#define S_    60
#define T_    20

typedef __attribute__((ext_vector_type(16))) __bf16 v16bf;
typedef __attribute__((ext_vector_type(8)))  float  v8f;

union AFrag { v16bf v; unsigned u[8]; };

// fp32 -> bf16 (round to nearest even), as raw bits
static __device__ __forceinline__ unsigned short f2b(float f) {
    unsigned u = __builtin_bit_cast(unsigned, f);
    u += 0x7fffu + ((u >> 16) & 1u);
    return (unsigned short)(u >> 16);
}
static __device__ __forceinline__ unsigned packbf(float lo, float hi) {
    return (unsigned)f2b(lo) | ((unsigned)f2b(hi) << 16);
}
static __device__ __forceinline__ float sigmoidf(float x) {
    return 1.0f / (1.0f + __expf(-x));
}

// Fragment loaders (ISA 7.12.2 VGPR layouts, wave32):
//  A (16x32 bf16): lane l: M=l&15, half=l>>4; VGPR p<4: K=2p+8*half; p>=4: K=2p+8+8*half (pairs)
//  B from row-major W[N,K]: lane l: N=l&15, half=l>>4; VGPR p<4: K=2p+8*half; p>=4: K=2p+8+8*half (pairs)
static __device__ __forceinline__ v16bf load_a_bf16(const unsigned short* __restrict__ arow,
                                                    int kb, int half) {
    AFrag a;
#pragma unroll
    for (int p = 0; p < 8; ++p) {
        const int ka = kb + 2 * p + ((p >= 4) ? 8 : 0) + 8 * half;
        a.u[p] = *(const unsigned*)(arow + ka);
    }
    return a.v;
}
static __device__ __forceinline__ v16bf load_a_f32(const float* __restrict__ arow,
                                                   int kb, int half) {
    AFrag a;
#pragma unroll
    for (int p = 0; p < 8; ++p) {
        const int ka = kb + 2 * p + ((p >= 4) ? 8 : 0) + 8 * half;
        a.u[p] = packbf(arow[ka], arow[ka + 1]);
    }
    return a.v;
}
static __device__ __forceinline__ v16bf load_b(const unsigned short* __restrict__ wrow,
                                               int kb, int half) {
    AFrag b;
#pragma unroll
    for (int p = 0; p < 8; ++p) {
        const int kw = kb + 2 * p + ((p >= 4) ? 8 : 0) + 8 * half;
        b.u[p] = *(const unsigned*)(wrow + kw);
    }
    return b.v;
}

// Tiled GEMM:  C[M,N] = A0[M,K0]*W0[N,K0]^T + bias0 (+ A1[M,K1]*W1[N,K1]^T + bias1)
// One wave computes a 16 x (NT*16) strip: the A fragment is loaded once per K-strip and
// reused for NT WMMAs (register blocking -> 4x fewer A loads, higher wmma:vmem ratio).
// All guards are wave-uniform so EXEC is all-ones at every WMMA (ISA 7.12 requirement).
template <bool A0_IS_F32, int NT>
__global__ __launch_bounds__(256)
void wmma_gemm(const void* __restrict__ A0v,
               const unsigned short* __restrict__ W0,
               const float* __restrict__ bias0,
               const unsigned short* __restrict__ A1,
               const unsigned short* __restrict__ W1,
               const float* __restrict__ bias1,
               float* __restrict__ Cout,
               int Mrows, int Ncols, int K0, int K1)
{
    const int tilesN = Ncols >> 4;
    const int wavesN = tilesN / NT;
    const int totalWaves = (Mrows >> 4) * wavesN;
    const int wave = blockIdx.x * (blockDim.x >> 5) + (threadIdx.x >> 5);
    if (wave >= totalWaves) return;                 // wave-uniform
    const int tileM  = wave / wavesN;
    const int tileN0 = (wave % wavesN) * NT;
    const int lane = threadIdx.x & 31;
    const int r    = lane & 15;
    const int half = lane >> 4;

    v8f acc[NT];
#pragma unroll
    for (int j = 0; j < NT; ++j) {
        const int col = (tileN0 + j) * 16 + r;
        float bias = bias0[col];
        if (bias1) bias += bias1[col];
#pragma unroll
        for (int i = 0; i < 8; ++i) acc[j][i] = bias;
    }

    const int rowA = tileM * 16 + r;
    // pair 0
    {
        const float*          arf = (const float*)A0v          + (size_t)rowA * K0;
        const unsigned short* arb = (const unsigned short*)A0v + (size_t)rowA * K0;
        for (int kb = 0; kb < K0; kb += 32) {
            __builtin_prefetch((A0_IS_F32 ? (const void*)(arf + kb + 32)
                                          : (const void*)(arb + kb + 32)), 0, 1);
            v16bf a;
            if (A0_IS_F32) a = load_a_f32(arf, kb, half);
            else           a = load_a_bf16(arb, kb, half);
#pragma unroll
            for (int j = 0; j < NT; ++j) {
                const unsigned short* wrow = W0 + (size_t)((tileN0 + j) * 16 + r) * K0;
                v16bf bfr = load_b(wrow, kb, half);
                acc[j] = __builtin_amdgcn_wmma_f32_16x16x32_bf16(
                    false, a, false, bfr, (short)0, acc[j], false, false);
            }
        }
    }
    // optional pair 1 (bf16 A)
    if (A1) {
        const unsigned short* arb = A1 + (size_t)rowA * K1;
        for (int kb = 0; kb < K1; kb += 32) {
            __builtin_prefetch(arb + kb + 32, 0, 1);
            v16bf a = load_a_bf16(arb, kb, half);
#pragma unroll
            for (int j = 0; j < NT; ++j) {
                const unsigned short* wrow = W1 + (size_t)((tileN0 + j) * 16 + r) * K1;
                v16bf bfr = load_b(wrow, kb, half);
                acc[j] = __builtin_amdgcn_wmma_f32_16x16x32_bf16(
                    false, a, false, bfr, (short)0, acc[j], false, false);
            }
        }
    }

    // D layout: VGPR r8 -> row tileM*16 + r8 + 8*half, col = tile col + (lane&15)
#pragma unroll
    for (int j = 0; j < NT; ++j) {
        const int col = (tileN0 + j) * 16 + r;
#pragma unroll
        for (int r8 = 0; r8 < 8; ++r8)
            Cout[(size_t)(tileM * 16 + r8 + 8 * half) * Ncols + col] = acc[j][r8];
    }
}

// fp32 -> bf16 bulk conversion (weights, once per launch)
__global__ void f2b_kernel(const float* __restrict__ src, unsigned short* __restrict__ dst, int n)
{
    int i = blockIdx.x * blockDim.x + threadIdx.x;
    if (i < n) dst[i] = f2b(src[i]);
}

// x = bf16(emb[tokens[:, t]])
__global__ void gather_embed(const int* __restrict__ tokens, const float* __restrict__ emb,
                             unsigned short* __restrict__ xb, int t)
{
    int i = blockIdx.x * blockDim.x + threadIdx.x;   // B_*H_ threads
    int b = i >> 9;                                   // /H_
    int h = i & (H_ - 1);
    int tok = tokens[b * T_ + t];
    xb[i] = f2b(emb[(size_t)tok * H_ + h]);
}

__global__ void init_state(float* __restrict__ hx, float* __restrict__ cx,
                           unsigned short* __restrict__ hxb)
{
    int i = blockIdx.x * blockDim.x + threadIdx.x;   // B_*H_ threads
    hx[i] = 0.0f; cx[i] = 0.0f; hxb[i] = 0;
}

// One block per (m,b): score->softmax->ctx, writes ctx as bf16 ready for the Wfh GEMM.
// hp row + w2 are staged into LDS with gfx1250 async-to-LDS DMA (ASYNCcnt-tracked).
__global__ __launch_bounds__(256)
void attn_kernel(const float* __restrict__ fproj, const float* __restrict__ hp,
                 const float* __restrict__ w2, const float* __restrict__ feats,
                 unsigned short* __restrict__ ctxb)
{
    const int mb = blockIdx.x;           // 0 .. M_*B_-1
    const int m  = mb / B_;
    const int b  = mb % B_;
    const int tid  = threadIdx.x;
    const int wave = tid >> 5;
    const int lane = tid & 31;

    __shared__ float sh_hp[MID_];
    __shared__ float sh_w2[MID_];
    __shared__ float sh_att[64];

    // Async global->LDS staging (flat shared-aperture pointer: low 32 bits = LDS byte addr)
    {
        unsigned           lds0 = (unsigned)(size_t)&sh_hp[tid];
        unsigned long long ga0  = (unsigned long long)(size_t)(hp + b * MID_ + tid);
        asm volatile("global_load_async_to_lds_b32 %0, %1, off"
                     :: "v"(lds0), "v"(ga0) : "memory");
        unsigned           lds1 = (unsigned)(size_t)&sh_w2[tid];
        unsigned long long ga1  = (unsigned long long)(size_t)(w2 + tid);
        asm volatile("global_load_async_to_lds_b32 %0, %1, off"
                     :: "v"(lds1), "v"(ga1) : "memory");
        asm volatile("s_wait_asynccnt 0" ::: "memory");
    }
    __syncthreads();

    const float* fp = fproj + (size_t)mb * S_ * MID_;
    for (int s = wave; s < S_; s += 8) {
        float part = 0.0f;
        const float* fps = fp + s * MID_;
#pragma unroll
        for (int kk = lane; kk < MID_; kk += 32)
            part += tanhf(fps[kk] + sh_hp[kk]) * sh_w2[kk];
#pragma unroll
        for (int off = 16; off > 0; off >>= 1)
            part += __shfl_down(part, off, 32);
        if (lane == 0) sh_att[s] = part;
    }
    __syncthreads();

    if (tid == 0) {   // S_=60: serial softmax normalization is negligible
        float mx = -1e30f;
        for (int s = 0; s < S_; ++s) mx = fmaxf(mx, sh_att[s]);
        float sum = 0.0f;
        for (int s = 0; s < S_; ++s) { float e = __expf(sh_att[s] - mx); sh_att[s] = e; sum += e; }
        float inv = 1.0f / sum;
        for (int s = 0; s < S_; ++s) sh_att[s] *= inv;
    }
    __syncthreads();

    const float* fb = feats + (size_t)mb * S_ * H_;
    for (int h = tid; h < H_; h += 256) {
        float acc = 0.0f;
        for (int s = 0; s < S_; ++s)
            acc += sh_att[s] * fb[s * H_ + h];
        ctxb[(size_t)b * (M_ * H_) + m * H_ + h] = f2b(acc);   // concat [B, M*H]
    }
}

// Fused gate nonlinearity + state update; also emits next-step h in bf16.
__global__ void lstm_pointwise(const float* __restrict__ gates, const float* __restrict__ tmp3,
                               float* __restrict__ hx, float* __restrict__ cx,
                               unsigned short* __restrict__ hxb,
                               float* __restrict__ out, int t)
{
    int i = blockIdx.x * blockDim.x + threadIdx.x;   // B_*H_ threads
    int b = i >> 9;
    int h = i & (H_ - 1);
    const float* g  = gates + (size_t)b * 4 * H_;
    const float* tm = tmp3  + (size_t)b * 3 * H_;
    float ig = g[h], fg = g[H_ + h], gg = g[2 * H_ + h], og = g[3 * H_ + h];
    float ti = tm[h], tf = tm[H_ + h], to = tm[2 * H_ + h];
    float I = sigmoidf(ig + ti);
    float F = sigmoidf(fg + tf);
    float G = tanhf(gg);
    float O = sigmoidf(og + to);
    float cy = F * cx[i] + I * G;
    float hy = O * tanhf(cy);
    cx[i] = cy;
    hx[i] = hy;
    hxb[i] = f2b(hy);
    out[(size_t)b * T_ * H_ + (size_t)t * H_ + h] = hy;
}

extern "C" void kernel_launch(void* const* d_in, const int* in_sizes, int n_in,
                              void* d_out, int out_size, void* d_ws, size_t ws_size,
                              hipStream_t stream)
{
    (void)in_sizes; (void)n_in; (void)out_size; (void)ws_size;
    const float* feats  = (const float*)d_in[0];   // [M,B,S,H]
    const int*   tokens = (const int*)  d_in[1];   // [B,T]
    const float* emb    = (const float*)d_in[2];   // [V,H]
    const float* Wh     = (const float*)d_in[3];   // [MID,H]
    const float* bh     = (const float*)d_in[4];
    const float* Wf     = (const float*)d_in[5];   // [MID,H]
    const float* bf     = (const float*)d_in[6];
    const float* w2     = (const float*)d_in[7];   // [MID]
    const float* Wih    = (const float*)d_in[8];   // [4H,H]
    const float* bih    = (const float*)d_in[9];
    const float* Whh    = (const float*)d_in[10];  // [4H,H]
    const float* bhh    = (const float*)d_in[11];
    const float* Wfh    = (const float*)d_in[12];  // [3H,M*H]
    const float* bfh    = (const float*)d_in[13];
    float* out = (float*)d_out;                    // [B,T,H]

    // ---- workspace carve (all chunk sizes 256B-aligned), ~26.6 MB total ----
    char* p = (char*)d_ws;
    float*          fproj = (float*)p;          p += (size_t)M_ * B_ * S_ * MID_ * 4;  // 15.7MB
    unsigned short* WfB   = (unsigned short*)p; p += (size_t)MID_ * H_ * 2;
    unsigned short* WhB   = (unsigned short*)p; p += (size_t)MID_ * H_ * 2;
    unsigned short* WihB  = (unsigned short*)p; p += (size_t)4 * H_ * H_ * 2;
    unsigned short* WhhB  = (unsigned short*)p; p += (size_t)4 * H_ * H_ * 2;
    unsigned short* WfhB  = (unsigned short*)p; p += (size_t)3 * H_ * M_ * H_ * 2;
    float*          hx    = (float*)p;          p += (size_t)B_ * H_ * 4;
    float*          cx    = (float*)p;          p += (size_t)B_ * H_ * 4;
    unsigned short* hxb   = (unsigned short*)p; p += (size_t)B_ * H_ * 2;
    unsigned short* xb    = (unsigned short*)p; p += (size_t)B_ * H_ * 2;
    float*          hp    = (float*)p;          p += (size_t)B_ * MID_ * 4;
    unsigned short* ctxb  = (unsigned short*)p; p += (size_t)B_ * M_ * H_ * 2;
    float*          gates = (float*)p;          p += (size_t)B_ * 4 * H_ * 4;
    float*          tmp3  = (float*)p;          p += (size_t)B_ * 3 * H_ * 4;

    // ---- weights -> bf16 (once per launch; graph-capture safe) ----
    auto cvt = [&](const float* s, unsigned short* d, int n) {
        f2b_kernel<<<(n + 255) / 256, 256, 0, stream>>>(s, d, n);
    };
    cvt(Wf,  WfB,  MID_ * H_);
    cvt(Wh,  WhB,  MID_ * H_);
    cvt(Wih, WihB, 4 * H_ * H_);
    cvt(Whh, WhhB, 4 * H_ * H_);
    cvt(Wfh, WfhB, 3 * H_ * M_ * H_);

    // ---- hoisted fproj = feats @ Wf^T + bf : [15360,256], K=512 ----
    {
        const int Mr = M_ * B_ * S_, Nc = MID_;
        const int waves = (Mr / 16) * ((Nc / 16) / 4);     // 3840 waves
        wmma_gemm<true, 4><<<waves / 8, 256, 0, stream>>>(
            (const void*)feats, WfB, bf, nullptr, nullptr, nullptr,
            fproj, Mr, Nc, H_, 0);
    }

    init_state<<<(B_ * H_) / 256, 256, 0, stream>>>(hx, cx, hxb);

    // ---- sequential decode ----
    for (int t = 0; t < T_; ++t) {
        gather_embed<<<(B_ * H_) / 256, 256, 0, stream>>>(tokens, emb, xb, t);

        // hp = h @ Wh^T + bh : [128,256], K=512 -> 32 waves
        wmma_gemm<false, 4><<<(B_ / 16) * ((MID_ / 16) / 4) / 8, 256, 0, stream>>>(
            (const void*)hxb, WhB, bh, nullptr, nullptr, nullptr,
            hp, B_, MID_, H_, 0);

        // attention: score/softmax/ctx (fused), ctx emitted bf16 [B, M*H]
        attn_kernel<<<M_ * B_, 256, 0, stream>>>(fproj, hp, w2, feats, ctxb);

        // gates = x@Wih^T + bih + h@Whh^T + bhh : [128,2048] -> 256 waves
        wmma_gemm<false, 4><<<(B_ / 16) * ((4 * H_ / 16) / 4) / 8, 256, 0, stream>>>(
            (const void*)xb, WihB, bih, hxb, WhhB, bhh,
            gates, B_, 4 * H_, H_, H_);

        // tmp = ctx@Wfh^T + bfh : [128,1536], K=1024 -> 192 waves
        wmma_gemm<false, 4><<<(B_ / 16) * ((3 * H_ / 16) / 4) / 8, 256, 0, stream>>>(
            (const void*)ctxb, WfhB, bfh, nullptr, nullptr, nullptr,
            tmp3, B_, 3 * H_, M_ * H_, 0);

        lstm_pointwise<<<(B_ * H_) / 256, 256, 0, stream>>>(
            gates, tmp3, hx, cx, hxb, out, t);
    }
}
